// DeformBottleneckBlock_2267742732697
// MI455X (gfx1250) — matrix operand order repaired and checked by reading, not compiled, not run on hardware
//
#include <hip/hip_runtime.h>
#include <hip/hip_bf16.h>

// ---------------------------------------------------------------------------
// DeformBottleneckBlock for MI455X (gfx1250), compile-only tuned.
// GEMMs: v_wmma_f32_16x16x32_bf16, LDS double-buffered via
// global_load_async_to_lds_b128 (ASYNCcnt) for the B tiles.
// ---------------------------------------------------------------------------

typedef __attribute__((ext_vector_type(16))) __bf16 v16bf;
typedef __attribute__((ext_vector_type(8)))  float  v8f;

static constexpr int Nn  = 4;
static constexpr int CIN = 1024;
static constexpr int Hh  = 56;
static constexpr int Ww  = 56;
static constexpr int CB  = 256;
static constexpr int Kk  = 9;
static constexpr int HW  = Hh * Ww;        // 3136
static constexpr int NP  = Nn * HW;        // 12544 pixels (multiple of 64)
static constexpr int K2  = CB * Kk;        // 2304

__device__ __forceinline__ unsigned short f2bf(float f) {
    unsigned int u = __float_as_uint(f);
    u += 0x7fffu + ((u >> 16) & 1u);       // round-to-nearest-even
    return (unsigned short)(u >> 16);
}
__device__ __forceinline__ float bf2f(unsigned short h) {
    return __uint_as_float(((unsigned int)h) << 16);
}

// ---- x (NCHW f32) -> pixel-major bf16 [pixel][CIN] --------------------------
__global__ void pack_x_kernel(const float* __restrict__ x,
                              unsigned short* __restrict__ xT) {
    long i = (long)blockIdx.x * blockDim.x + threadIdx.x;
    const long total = (long)Nn * CIN * HW;
    if (i >= total) return;
    int  hw = (int)(i % HW);
    long t  = i / HW;
    int  ci = (int)(t % CIN);
    int  n  = (int)(t / CIN);
    xT[(long)(n * HW + hw) * CIN + ci] = f2bf(x[i]);
}

// ---- generic f32 -> bf16 cast (w1, w3 are already row-major [M][K]) --------
__global__ void cast_bf16_kernel(const float* __restrict__ in,
                                 unsigned short* __restrict__ out, long n) {
    long i = (long)blockIdx.x * blockDim.x + threadIdx.x;
    if (i < n) out[i] = f2bf(in[i]);
}

// ---- w2 (oc, c, 3, 3) f32 -> A2 bf16 [oc][k*CB + c] ------------------------
__global__ void pack_w2_kernel(const float* __restrict__ w2,
                               unsigned short* __restrict__ A2) {
    long i = (long)blockIdx.x * blockDim.x + threadIdx.x;
    const long total = (long)CB * K2;
    if (i >= total) return;
    int oc = (int)(i / K2);
    int rr = (int)(i % K2);
    int k  = rr / CB;
    int c  = rr % CB;
    A2[i] = f2bf(w2[((long)oc * CB + c) * Kk + k]);
}

// ---------------------------------------------------------------------------
// bf16 WMMA GEMM:  C[m, pix] = sum_k A[m][k] * B[pix][k]   (A row-major MxK,
// B pixel-major "B^T" so both operands are K-contiguous -> NT GEMM).
// Block tile 128(M) x 64(N), 8 waves arranged 4x2, each wave 2x2 16x16 frags.
// B tiles double-buffered in LDS via global_load_async_to_lds_b128.
// MODE 0: out = relu(C + bias) -> bf16 pixel-major [pix][M]
// MODE 1: out = relu(C + bias + resid) -> f32 NCHW (final output)
// ---------------------------------------------------------------------------
template <int MODE>
__global__ __launch_bounds__(256) void gemm_bf16_kernel(
    const unsigned short* __restrict__ A,
    const unsigned short* __restrict__ B,
    const float* __restrict__ bias, int M, int K,
    unsigned short* __restrict__ outB,
    float* __restrict__ outF, const float* __restrict__ resid) {

    // two 4KB buffers: 64 pixels x 32 K bf16 each
    __shared__ __align__(16) unsigned short ldsB[2][64 * 32];

    const int tid  = threadIdx.x;
    const int lane = tid & 31;
    const int wave = tid >> 5;
    const int wm   = wave & 3;   // 4 waves along M
    const int wn   = wave >> 2;  // 2 waves along N
    const int n0   = blockIdx.x * 64;
    const int m0   = blockIdx.y * 128;

    __builtin_amdgcn_s_wait_tensorcnt(0); // gfx1250 split-counter path

    v8f acc[2][2] = {};

    // A fragment lane layout (16-bit A 16x32): lane<16 holds K {b..b+7, b+16..b+23}
    const int arow = lane & 15;
    const int akb  = (lane < 16) ? 0 : 8;
    // B fragment lane layout (32x16): lane<16 holds K {b..b+7, b+16..b+23} of one column
    const int bp   = lane & 15;
    const int bkb  = (lane < 16) ? 0 : 8;

    // each of the 256 threads asynchronously copies 16B of the 4KB B tile
    const int sp = tid >> 2, sq = tid & 3;
    auto stage_async = [&](int kk, int buf) {
        const unsigned short* src = B + (long)(n0 + sp) * K + kk + sq * 8;
        unsigned int dst =
            (unsigned int)(uintptr_t)&ldsB[buf][sp * 32 + sq * 8];
        asm volatile("global_load_async_to_lds_b128 %0, %1, off"
                     :: "v"(dst), "v"(src) : "memory");
    };

    const int nsteps = K / 32;
    stage_async(0, 0);

    for (int s = 0; s < nsteps; ++s) {
        const int kk = s * 32;
        if (s + 1 < nsteps) {
            stage_async(kk + 32, (s + 1) & 1);
            // loads complete in order: <=1 outstanding => tile s is resident
            asm volatile("s_wait_asynccnt 0x1" ::: "memory");
        } else {
            asm volatile("s_wait_asynccnt 0x0" ::: "memory");
        }
        __syncthreads(); // all waves' async portions of tile s visible

        const unsigned short* lbuf = ldsB[s & 1];

        union Frag { float4 f4[2]; v16bf v; } af[2], bfr[2];
#pragma unroll
        for (int mt = 0; mt < 2; ++mt) {
            const unsigned short* ap =
                A + (long)(m0 + wm * 32 + mt * 16 + arow) * K + kk + akb;
            af[mt].f4[0] = *reinterpret_cast<const float4*>(ap);
            af[mt].f4[1] = *reinterpret_cast<const float4*>(ap + 16);
            __builtin_prefetch(ap + 32, 0, 3);
        }
#pragma unroll
        for (int nt = 0; nt < 2; ++nt) {
            const unsigned short* lp = &lbuf[(wn * 32 + nt * 16 + bp) * 32 + bkb];
            bfr[nt].f4[0] = *reinterpret_cast<const float4*>(lp);
            bfr[nt].f4[1] = *reinterpret_cast<const float4*>(lp + 16);
        }
#pragma unroll
        for (int mt = 0; mt < 2; ++mt)
#pragma unroll
            for (int nt = 0; nt < 2; ++nt)
                acc[mt][nt] = __builtin_amdgcn_wmma_f32_16x16x32_bf16(
                    false, af[mt].v, false, bfr[nt].v,
                    (short)0, acc[mt][nt], false, false);

        // protect buffer s&1 == buffer overwritten by next iter's issue
        __syncthreads();
    }

    // Epilogue. C layout: vgpr i -> M = i + (lane<16?0:8); N = lane%16.
#pragma unroll
    for (int mt = 0; mt < 2; ++mt)
#pragma unroll
        for (int nt = 0; nt < 2; ++nt)
#pragma unroll
            for (int i = 0; i < 8; ++i) {
                int m   = m0 + wm * 32 + mt * 16 + i + ((lane < 16) ? 0 : 8);
                int pix = n0 + wn * 32 + nt * 16 + (lane & 15);
                float v = acc[mt][nt][i] + bias[m];
                if (MODE == 0) {
                    v = v > 0.f ? v : 0.f;
                    outB[(long)pix * M + m] = f2bf(v);
                } else {
                    int  n  = pix / HW, hw = pix % HW;
                    long idx = ((long)n * M + m) * HW + hw;
                    v += resid[idx];
                    v = v > 0.f ? v : 0.f;
                    outF[idx] = v;
                }
            }
}

// ---- offset conv: 3x3, 18 out channels, SAME zero pad, input NHWC bf16 ----
__global__ void offset_conv_kernel(const unsigned short* __restrict__ r,
                                   const float* __restrict__ wof,
                                   const float* __restrict__ bof,
                                   float* __restrict__ off) {
    int gid = blockIdx.x * blockDim.x + threadIdx.x;
    const int total = Nn * 18 * HW;
    if (gid >= total) return;
    int hw = gid % HW;
    int t  = gid / HW;
    int oc = t % 18;
    int n  = t / 18;
    int h  = hw / Ww, w = hw % Ww;

    float acc = bof[oc];
    for (int ki = 0; ki < 3; ++ki) {
        int y = h + ki - 1;
        if (y < 0 || y >= Hh) continue;
        for (int kj = 0; kj < 3; ++kj) {
            int x = w + kj - 1;
            if (x < 0 || x >= Ww) continue;
            const unsigned short* rp = r + ((long)(n * HW + y * Ww + x)) * CB;
            const float* wp = wof + (long)oc * K2 + ki * 3 + kj; // stride 9 per c
            for (int c = 0; c < CB; c += 4) {
                uint2 pk = *reinterpret_cast<const uint2*>(rp + c);
                acc += bf2f((unsigned short)(pk.x & 0xffffu)) * wp[(c + 0) * 9];
                acc += bf2f((unsigned short)(pk.x >> 16))     * wp[(c + 1) * 9];
                acc += bf2f((unsigned short)(pk.y & 0xffffu)) * wp[(c + 2) * 9];
                acc += bf2f((unsigned short)(pk.y >> 16))     * wp[(c + 3) * 9];
            }
        }
    }
    off[gid] = acc; // layout (n, 18, h, w) == (n, k, {dy,dx}, h, w)
}

// ---- bilinear gather -> im2col bf16 [pix][k*CB + c] ------------------------
__global__ __launch_bounds__(64) void sample_kernel(
    const unsigned short* __restrict__ r, const float* __restrict__ off,
    unsigned short* __restrict__ col) {
    int g   = blockIdx.x;          // Nn*HW*9 blocks
    int k   = g % Kk;
    int pix = g / Kk;
    int n   = pix / HW, hw = pix % HW;
    int h   = hw / Ww,  w  = hw % Ww;

    float dy = off[((long)n * 18 + k * 2 + 0) * HW + hw];
    float dx = off[((long)n * 18 + k * 2 + 1) * HW + hw];
    float ys = (float)h + (float)(k / 3 - 1) + dy;
    float xs = (float)w + (float)(k % 3 - 1) + dx;
    float y0 = floorf(ys), x0 = floorf(xs);
    float wy1 = ys - y0, wy0 = 1.f - wy1;
    float wx1 = xs - x0, wx0 = 1.f - wx1;

    float wgt[4] = { wy0 * wx0, wy0 * wx1, wy1 * wx0, wy1 * wx1 };
    long  base[4];
#pragma unroll
    for (int t = 0; t < 4; ++t) {
        int yi = (int)y0 + (t >> 1);
        int xi = (int)x0 + (t & 1);
        bool valid = (yi >= 0) && (yi < Hh) && (xi >= 0) && (xi < Ww);
        if (!valid) wgt[t] = 0.f;
        int yc = min(max(yi, 0), Hh - 1);
        int xc = min(max(xi, 0), Ww - 1);
        base[t] = ((long)n * HW + yc * Ww + xc) * CB;
    }

    int c = threadIdx.x * 4;       // 64 threads x 4 channels = 256
    float v0 = 0.f, v1 = 0.f, v2 = 0.f, v3 = 0.f;
#pragma unroll
    for (int t = 0; t < 4; ++t) {
        uint2 pk = *reinterpret_cast<const uint2*>(r + base[t] + c);
        float wv = wgt[t];
        v0 += wv * bf2f((unsigned short)(pk.x & 0xffffu));
        v1 += wv * bf2f((unsigned short)(pk.x >> 16));
        v2 += wv * bf2f((unsigned short)(pk.y & 0xffffu));
        v3 += wv * bf2f((unsigned short)(pk.y >> 16));
    }
    unsigned short* dst = col + (long)pix * K2 + k * CB + c;
    dst[0] = f2bf(v0); dst[1] = f2bf(v1);
    dst[2] = f2bf(v2); dst[3] = f2bf(v3);
}

// ---------------------------------------------------------------------------
extern "C" void kernel_launch(void* const* d_in, const int* in_sizes, int n_in,
                              void* d_out, int out_size, void* d_ws, size_t ws_size,
                              hipStream_t stream) {
    (void)in_sizes; (void)n_in; (void)out_size; (void)ws_size;

    const float* x   = (const float*)d_in[0];
    const float* w1  = (const float*)d_in[1];
    const float* b1  = (const float*)d_in[2];
    const float* wof = (const float*)d_in[3];
    const float* bof = (const float*)d_in[4];
    const float* w2  = (const float*)d_in[5];
    const float* b2  = (const float*)d_in[6];
    const float* w3  = (const float*)d_in[7];
    const float* b3  = (const float*)d_in[8];
    float* out = (float*)d_out;

    char*  ws = (char*)d_ws;
    size_t p  = 0;
    auto alloc = [&](size_t bytes) -> void* {
        void* r = ws + p;
        p = (p + bytes + 255) & ~(size_t)255;
        return r;
    };

    // col (57.8MB) reuses the xT region: xT only live during gemm1, col after.
    unsigned short* colx = (unsigned short*)alloc((size_t)NP * K2 * 2);
    unsigned short* xT   = colx;
    unsigned short* rB   = (unsigned short*)alloc((size_t)NP * CB * 2);
    unsigned short* o2B  = (unsigned short*)alloc((size_t)NP * CB * 2);
    float*          offb = (float*)alloc((size_t)Nn * 18 * HW * 4);
    unsigned short* A1   = (unsigned short*)alloc((size_t)CB * CIN * 2);
    unsigned short* A2   = (unsigned short*)alloc((size_t)CB * K2 * 2);
    unsigned short* A3   = (unsigned short*)alloc((size_t)CIN * CB * 2);

    const long totX = (long)Nn * CIN * HW;
    pack_x_kernel<<<(int)((totX + 255) / 256), 256, 0, stream>>>(x, xT);
    cast_bf16_kernel<<<(CB * CIN + 255) / 256, 256, 0, stream>>>(w1, A1, (long)CB * CIN);
    cast_bf16_kernel<<<(CIN * CB + 255) / 256, 256, 0, stream>>>(w3, A3, (long)CIN * CB);
    pack_w2_kernel<<<(CB * K2 + 255) / 256, 256, 0, stream>>>(w2, A2);

    // conv1: r = relu(w1 @ xT + b1), bf16 NHWC
    gemm_bf16_kernel<0><<<dim3(NP / 64, CB / 128), 256, 0, stream>>>(
        A1, xT, b1, CB, CIN, rB, nullptr, nullptr);

    // offsets + deformable im2col
    offset_conv_kernel<<<(Nn * 18 * HW + 255) / 256, 256, 0, stream>>>(rB, wof, bof, offb);
    sample_kernel<<<Nn * HW * Kk, 64, 0, stream>>>(rB, offb, colx);

    // deform matmul: out2 = relu(w2 @ col + b2), bf16 NHWC
    gemm_bf16_kernel<0><<<dim3(NP / 64, CB / 128), 256, 0, stream>>>(
        A2, colx, b2, CB, K2, o2B, nullptr, nullptr);

    // conv3 + residual + relu -> final NCHW f32
    gemm_bf16_kernel<1><<<dim3(NP / 64, CIN / 128), 256, 0, stream>>>(
        A3, o2B, b3, CIN, CB, nullptr, out, x);
}
